// Attention_3599182594722
// MI455X (gfx1250) — matrix operand order repaired and checked by reading, not compiled, not run on hardware
//
#include <hip/hip_runtime.h>
#include <hip/hip_bf16.h>
#include <cstdint>
#include <cstddef>

// ---------------------------------------------------------------------------
// Shapes (fixed by the reference)
// ---------------------------------------------------------------------------
#define NSEQ    4096
#define HIDDEN  1024
#define NH      16
#define HD      64
#define QKV_DIM 3072   // 3 * NH * HD

typedef __attribute__((ext_vector_type(16))) __bf16    v16bf;
typedef __attribute__((ext_vector_type(8)))  float     v8f;
typedef __attribute__((ext_vector_type(4)))  uint32_t  u32x4;
typedef __attribute__((ext_vector_type(8)))  uint32_t  u32x8;

union AFrag {           // 16x32 bf16 A/B operand = 8 VGPRs = 32 bytes/lane
    v16bf v;
    uint4 q[2];
};

static __device__ __forceinline__ v8f wmma_bf16(const AFrag& a, const AFrag& b, v8f c) {
    return __builtin_amdgcn_wmma_f32_16x16x32_bf16(
        /*neg_a=*/false, a.v, /*neg_b=*/false, b.v,
        /*c_mod=*/(short)0, c, /*reuse_a=*/false, /*reuse_b=*/false);
}

// ---------------------------------------------------------------------------
// Tensor Data Mover: async 2D tile load Global -> LDS (CDNA5 ISA-8).
// D# group0 (4 SGPRs) + group1 (8 SGPRs), 2-group form (2D tensor,
// VADDR2/VADDR3 = NULL). data_size = 2 bytes (bf16). Optional LDS row
// padding via pad_enable/pad_interval/pad_amount so the DMA itself lays
// out the bank-conflict-padded tile. Issue from ONE wave; completion via
// s_wait_tensorcnt 0 then a workgroup barrier.
// ---------------------------------------------------------------------------
static __device__ __forceinline__ void tdm_load_tile_2d_bf16(
        const void* gptr, unsigned lds_byte_off,
        unsigned tile_w, unsigned tile_h,          // elements / rows
        unsigned tensor_w, unsigned tensor_h,      // OOB extents (elements)
        unsigned stride_elems,                     // row stride (elements)
        unsigned pad_interval_code, unsigned pad_amount_code) {
    uint64_t ga = (uint64_t)(uintptr_t)gptr;
    u32x4 g0;
    g0[0] = 1u;                                                  // count=1, user D#
    g0[1] = lds_byte_off;                                        // lds_addr
    g0[2] = (uint32_t)ga;                                        // global_addr lo
    g0[3] = (uint32_t)((ga >> 32) & 0x01FFFFFFu) | (2u << 30);   // ga[56:32] | type=2
    u32x8 g1;
    g1[0] = (1u << 16)                                           // data_size = 2B
          | (1u << 20)                                           // pad_enable
          | (pad_interval_code << 22)
          | (pad_amount_code << 25);
    g1[1] = (tensor_w & 0xFFFFu) << 16;                          // tensor_dim0[15:0]
    g1[2] = (tensor_w >> 16) | ((tensor_h & 0xFFFFu) << 16);     // dim0 hi | dim1 lo
    g1[3] = (tensor_h >> 16) | (tile_w << 16);                   // dim1 hi | tile_dim0
    g1[4] = tile_h;                                              // tile_dim1 | tile_dim2=0
    g1[5] = stride_elems;                                        // dim0_stride[31:0]
    g1[6] = 0;                                                   // stride hi | dim1_stride lo
    g1[7] = 0;
    asm volatile("tensor_load_to_lds %0, %1" :: "s"(g0), "s"(g1) : "memory");
}

static __device__ __forceinline__ unsigned lds_offset_u32(const void* p) {
    return (unsigned)(uintptr_t)p;   // LDS aperture: addr[31:0] = LDS byte offset
}

// ---------------------------------------------------------------------------
// f32 -> bf16 conversion (x, w_qkv, w_out)
// ---------------------------------------------------------------------------
__global__ __launch_bounds__(256)
void attn_cvt_f32_bf16(const float* __restrict__ in, __bf16* __restrict__ out, int n) {
    int i = (blockIdx.x * 256 + threadIdx.x) * 4;
    if (i + 3 < n) {
        float4 f = *(const float4*)(in + i);
        out[i + 0] = (__bf16)f.x;
        out[i + 1] = (__bf16)f.y;
        out[i + 2] = (__bf16)f.z;
        out[i + 3] = (__bf16)f.w;
    }
}

// ---------------------------------------------------------------------------
// bf16 WMMA GEMM:  C[M][N] = A[M][K] @ B[K][N] + bias
// Block tile 128x128, BK=32, 256 threads = 8 waves (2x4), wave tile 64x32.
// A tile staged by TDM (padded rows, 40 bf16 stride); B tile staged
// transposed by the waves. OUT_F32 selects f32 vs bf16 output (no branches).
// ---------------------------------------------------------------------------
template <bool OUT_F32>
__global__ __launch_bounds__(256)
void attn_gemm_bf16_wmma(const __bf16* __restrict__ A, const __bf16* __restrict__ B,
                         const float* __restrict__ bias, void* __restrict__ Cout,
                         int M, int N, int K) {
    __shared__ __align__(16) __bf16 As[128][40];   // filled by TDM (pad 64B->+16B)
    __shared__ __align__(16) __bf16 Bt[128][40];   // B tile TRANSPOSED: [n][k]

    const int tid  = threadIdx.x;
    const int lane = tid & 31;
    const int wave = tid >> 5;
    const int half = lane >> 4;       // K-half selector for fragments
    const int ln   = lane & 15;       // row (A) / column (B,C) within 16
    const int wm   = wave >> 2;       // 0..1
    const int wn   = wave & 3;        // 0..3
    const int m0   = blockIdx.y * 128;
    const int n0   = blockIdx.x * 128;
    const unsigned asOff = lds_offset_u32(&As[0][0]);

    v8f acc[4][2];
    const v8f vzero = 0.0f;
#pragma unroll
    for (int mi = 0; mi < 4; ++mi)
#pragma unroll
        for (int ni = 0; ni < 2; ++ni) acc[mi][ni] = vzero;

    for (int kk = 0; kk < K; kk += 32) {
        // ---- TDM: A tile 128 rows x 32 bf16, row stride K, pad 64B->16B ----
        if (wave == 0) {
            tdm_load_tile_2d_bf16(A + (size_t)m0 * K + kk, asOff,
                                  /*tile_w=*/32, /*tile_h=*/128,
                                  /*tensor_w=*/(unsigned)K, /*tensor_h=*/0x100000u,
                                  /*stride=*/(unsigned)K,
                                  /*interval 16 DW=64B*/3u, /*amount 4 DW=16B*/3u);
        }
        // ---- stage B tile transposed (32 k x 128 n -> Bt[n][k]) ----
#pragma unroll
        for (int it = 0; it < 2; ++it) {
            int idx = tid + it * 256;          // 512 chunks of 8 bf16
            int kr = idx >> 4, cn = idx & 15;
            uint4 val = *(const uint4*)(B + (size_t)(kk + kr) * N + n0 + cn * 8);
            const __bf16* e = (const __bf16*)&val;
#pragma unroll
            for (int i = 0; i < 8; ++i) Bt[cn * 8 + i][kr] = e[i];
        }
        if (wave == 0) __builtin_amdgcn_s_wait_tensorcnt(0);
        __syncthreads();

        // ---- fragments from LDS (layouts per ISA 7.12.2) ----
        AFrag a[4], b[2];
#pragma unroll
        for (int mi = 0; mi < 4; ++mi) {
            int row = wm * 64 + mi * 16 + ln;
            a[mi].q[0] = *(const uint4*)(&As[row][half * 8]);        // K = half*8 .. +7
            a[mi].q[1] = *(const uint4*)(&As[row][16 + half * 8]);   // K = 16+half*8 .. +7
        }
#pragma unroll
        for (int ni = 0; ni < 2; ++ni) {
            int col = wn * 32 + ni * 16 + ln;
            b[ni].q[0] = *(const uint4*)(&Bt[col][half * 8]);        // K = half*8 .. +7
            b[ni].q[1] = *(const uint4*)(&Bt[col][16 + half * 8]);   // K = 16+half*8 .. +7
        }
#pragma unroll
        for (int mi = 0; mi < 4; ++mi)
#pragma unroll
            for (int ni = 0; ni < 2; ++ni)
                acc[mi][ni] = wmma_bf16(a[mi], b[ni], acc[mi][ni]);
        __syncthreads();   // tile consumed before next TDM / B staging
    }

    // ---- branch-free epilogue: bias + store (row = r + 8*half, col = ln) ----
#pragma unroll
    for (int ni = 0; ni < 2; ++ni) {
        const int col = n0 + wn * 32 + ni * 16 + ln;
        const float bv = bias[col];
#pragma unroll
        for (int mi = 0; mi < 4; ++mi) {
#pragma unroll
            for (int r = 0; r < 8; ++r) {
                const int row = m0 + wm * 64 + mi * 16 + half * 8 + r;
                const float val = acc[mi][ni][r] + bv;
                if constexpr (OUT_F32)
                    ((float*)Cout)[(size_t)row * N + col] = val;
                else
                    ((__bf16*)Cout)[(size_t)row * N + col] = (__bf16)val;
            }
        }
    }
}

// ---------------------------------------------------------------------------
// Flash attention over qkv (bf16, [4096][3072] row-major: q|k|v each 1024).
// Grid: (NSEQ/128, NH). 256 threads = 8 waves, each wave owns 16 query rows.
// K tile staged by TDM (padded); V tile staged transposed; online softmax;
// f32 accumulators; bf16 output [4096][1024].
// ---------------------------------------------------------------------------
__global__ __launch_bounds__(256)
void attn_flash_bf16_wmma(const __bf16* __restrict__ qkv, __bf16* __restrict__ attn) {
    __shared__ __align__(16) __bf16 kt[64][72];        // K tile: [key][d], TDM-filled
    __shared__ __align__(16) __bf16 vt[64][72];        // V tile transposed: [d][key]
    __shared__ __align__(16) __bf16 plds[8][16][72];   // per-wave P staging

    const int tid  = threadIdx.x;
    const int lane = tid & 31;
    const int wave = tid >> 5;
    const int half = lane >> 4;
    const int ln   = lane & 15;
    const int h    = blockIdx.y;
    const int q0   = blockIdx.x * 128;
    const unsigned ktOff = lds_offset_u32(&kt[0][0]);

    // ---- Q A-fragments straight from global (contiguous in d) ----
    AFrag qf[2];
    {
        const __bf16* qb = qkv + (size_t)(q0 + wave * 16 + ln) * QKV_DIM + h * HD;
#pragma unroll
        for (int kf = 0; kf < 2; ++kf) {
            qf[kf].q[0] = *(const uint4*)(qb + kf * 32 + half * 8);
            qf[kf].q[1] = *(const uint4*)(qb + kf * 32 + 16 + half * 8);
        }
    }

    v8f  o[4];
    float m_i[8], l_i[8];
    const v8f vzero = 0.0f;
#pragma unroll
    for (int di = 0; di < 4; ++di) o[di] = vzero;
#pragma unroll
    for (int r = 0; r < 8; ++r) { m_i[r] = -INFINITY; l_i[r] = 0.0f; }

    for (int t0 = 0; t0 < NSEQ; t0 += 64) {
        // ---- TDM: K tile 64 rows x 64 bf16, stride QKV_DIM, pad 128B->16B ----
        if (wave == 0) {
            tdm_load_tile_2d_bf16(qkv + (size_t)t0 * QKV_DIM + HIDDEN + h * HD, ktOff,
                                  /*tile_w=*/64, /*tile_h=*/64,
                                  /*tensor_w=*/QKV_DIM, /*tensor_h=*/0x100000u,
                                  /*stride=*/QKV_DIM,
                                  /*interval 32 DW=128B*/4u, /*amount 4 DW=16B*/3u);
        }
        // ---- stage V tile transposed ----
#pragma unroll
        for (int it = 0; it < 2; ++it) {
            int idx = tid + it * 256;          // 512 chunks of 8 bf16
            int kl = idx >> 3, cq = idx & 7;
            uint4 vv = *(const uint4*)(qkv + (size_t)(t0 + kl) * QKV_DIM + 2 * HIDDEN + h * HD + cq * 8);
            const __bf16* e = (const __bf16*)&vv;
#pragma unroll
            for (int i = 0; i < 8; ++i) vt[cq * 8 + i][kl] = e[i];
        }
        if (wave == 0) __builtin_amdgcn_s_wait_tensorcnt(0);
        __syncthreads();

        // ---- S = (Q Kt) * scale : 4 key subtiles x 2 k-steps over d ----
        v8f s[4];
#pragma unroll
        for (int ni = 0; ni < 4; ++ni) {
            s[ni] = vzero;
            int key = ni * 16 + ln;            // B-fragment column = key
#pragma unroll
            for (int kf = 0; kf < 2; ++kf) {
                AFrag kb;
                kb.q[0] = *(const uint4*)(&kt[key][kf * 32 + half * 8]);
                kb.q[1] = *(const uint4*)(&kt[key][kf * 32 + 16 + half * 8]);
                s[ni] = wmma_bf16(qf[kf], kb, s[ni]);
            }
            s[ni] *= 0.125f;                   // HEAD_DIM^-0.5
        }

        // ---- online softmax (rows live across 16 column lanes) ----
        float mnew[8], alpha[8];
#pragma unroll
        for (int r = 0; r < 8; ++r) {
            float mx = -INFINITY;
#pragma unroll
            for (int ni = 0; ni < 4; ++ni) mx = fmaxf(mx, s[ni][r]);
#pragma unroll
            for (int off = 8; off >= 1; off >>= 1)
                mx = fmaxf(mx, __shfl_xor(mx, off, 32));
            mnew[r]  = fmaxf(m_i[r], mx);
            alpha[r] = __expf(m_i[r] - mnew[r]);
            m_i[r]   = mnew[r];
        }
#pragma unroll
        for (int r = 0; r < 8; ++r) {
            float rs = 0.0f;
#pragma unroll
            for (int ni = 0; ni < 4; ++ni) {
                float p = __expf(s[ni][r] - mnew[r]);
                rs += p;
                plds[wave][half * 8 + r][ni * 16 + ln] = (__bf16)p;  // C-layout -> LDS
            }
#pragma unroll
            for (int off = 8; off >= 1; off >>= 1)
                rs += __shfl_xor(rs, off, 32);
            l_i[r] = l_i[r] * alpha[r] + rs;
        }
#pragma unroll
        for (int di = 0; di < 4; ++di)
#pragma unroll
            for (int r = 0; r < 8; ++r) o[di][r] *= alpha[r];

        // ---- reload P in A-layout (wave-local LDS, same-wave DS ordering) ----
        AFrag pf[2];
#pragma unroll
        for (int pk = 0; pk < 2; ++pk) {
            pf[pk].q[0] = *(const uint4*)(&plds[wave][ln][pk * 32 + half * 8]);
            pf[pk].q[1] = *(const uint4*)(&plds[wave][ln][pk * 32 + 16 + half * 8]);
        }

        // ---- O += P V : 4 d subtiles x 2 k-steps over keys ----
#pragma unroll
        for (int di = 0; di < 4; ++di) {
            int dcol = di * 16 + ln;           // B-fragment column = d
#pragma unroll
            for (int pk = 0; pk < 2; ++pk) {
                AFrag vb;
                vb.q[0] = *(const uint4*)(&vt[dcol][pk * 32 + half * 8]);
                vb.q[1] = *(const uint4*)(&vt[dcol][pk * 32 + 16 + half * 8]);
                o[di] = wmma_bf16(pf[pk], vb, o[di]);
            }
        }
        __syncthreads();   // tile consumed before next TDM / V staging
    }

    // ---- normalize and store bf16 attention output [4096][1024] ----
#pragma unroll
    for (int r = 0; r < 8; ++r) {
        float inv = 1.0f / l_i[r];
        int row = q0 + wave * 16 + half * 8 + r;
#pragma unroll
        for (int di = 0; di < 4; ++di)
            attn[(size_t)row * HIDDEN + h * HD + di * 16 + ln] = (__bf16)(o[di][r] * inv);
    }
}

// ---------------------------------------------------------------------------
// Host driver
// ---------------------------------------------------------------------------
extern "C" void kernel_launch(void* const* d_in, const int* in_sizes, int n_in,
                              void* d_out, int out_size, void* d_ws, size_t ws_size,
                              hipStream_t stream) {
    (void)in_sizes; (void)n_in; (void)out_size; (void)ws_size;
    const float* x     = (const float*)d_in[0];
    const float* w_qkv = (const float*)d_in[1];
    const float* b_qkv = (const float*)d_in[2];
    const float* w_out = (const float*)d_in[3];
    const float* b_out = (const float*)d_in[4];
    float* out = (float*)d_out;

    char* ws = (char*)d_ws;                                   // MiB offsets
    __bf16* x_bf    = (__bf16*)(ws);                          //  8 MiB  4096x1024
    __bf16* wqkv_bf = (__bf16*)(ws + ( 8ull << 20));          //  6 MiB  1024x3072
    __bf16* wout_bf = (__bf16*)(ws + (14ull << 20));          //  2 MiB  1024x1024
    __bf16* qkv_bf  = (__bf16*)(ws + (16ull << 20));          // 24 MiB  4096x3072
    __bf16* attn_bf = (__bf16*)(ws + (40ull << 20));          //  8 MiB  4096x1024

    attn_cvt_f32_bf16<<<(NSEQ * HIDDEN) / 1024, 256, 0, stream>>>(x, x_bf, NSEQ * HIDDEN);
    attn_cvt_f32_bf16<<<(HIDDEN * QKV_DIM) / 1024, 256, 0, stream>>>(w_qkv, wqkv_bf, HIDDEN * QKV_DIM);
    attn_cvt_f32_bf16<<<(HIDDEN * HIDDEN) / 1024, 256, 0, stream>>>(w_out, wout_bf, HIDDEN * HIDDEN);

    dim3 g1(QKV_DIM / 128, NSEQ / 128);   // qkv = x @ w_qkv + b_qkv  -> bf16
    attn_gemm_bf16_wmma<false><<<g1, 256, 0, stream>>>(x_bf, wqkv_bf, b_qkv,
                                                       (void*)qkv_bf,
                                                       NSEQ, QKV_DIM, HIDDEN);

    dim3 g2(NSEQ / 128, NH);              // flash attention -> attn_bf
    attn_flash_bf16_wmma<<<g2, 256, 0, stream>>>(qkv_bf, attn_bf);

    dim3 g3(HIDDEN / 128, NSEQ / 128);    // out = attn @ w_out + b_out -> f32
    attn_gemm_bf16_wmma<true><<<g3, 256, 0, stream>>>(attn_bf, wout_bf, b_out,
                                                      (void*)out,
                                                      NSEQ, HIDDEN, HIDDEN);
}
